// AttentionBlock_76355928588710
// MI455X (gfx1250) — matrix operand order repaired and checked by reading, not compiled, not run on hardware
//
#include <hip/hip_runtime.h>
#include <hip/hip_bf16.h>

// ---------------------------------------------------------------------------
// Multi-head attention block for MI455X (gfx1250, wave32, WMMA bf16).
//   B=4, S=2048, E=1024, H=16, D=64.
// All GEMMs through v_wmma_f32_16x16x32_bf16 (fp32 accumulate).
// Flash-attention online softmax so scores never hit HBM.
// Global->LDS staging uses the CDNA5 async path
// (global_load_async_to_lds_b128 + ASYNCcnt) with double-buffered LDS:
// one barrier per K-step and no VGPR round-trip for tile data.
// B-fragments are pre-gathered so the 4 WMMAs per step issue back-to-back.
// ---------------------------------------------------------------------------

#define EMBED 1024
#define HEADS 16
#define HDIM  64
#define SEQ   2048
#define BATCH 4
#define TOKENS (BATCH * SEQ)   // 8192

typedef __attribute__((ext_vector_type(16))) __bf16 v16bf;
typedef __attribute__((ext_vector_type(8)))  float  v8f;

union Frag16 { v16bf v; unsigned int u[8]; };

__device__ __forceinline__ unsigned short f2bf(float f) {
  unsigned int u = __float_as_uint(f);
  u += 0x7fffu + ((u >> 16) & 1u);       // round-to-nearest-even
  return (unsigned short)(u >> 16);
}

// A-matrix (16x32 bf16) per-lane K base for VGPR v (ISA 7.12.2):
// lanes 0-15: V0..3 -> K{0..7}, V4..7 -> K{16..23}; lanes 16-31: +8.
__device__ __forceinline__ int a_kbase(int v, int hi) {
  return (v < 4 ? 2 * v : 16 + 2 * (v - 4)) + hi * 8;
}

__device__ __forceinline__ v8f wmma_bf16(const Frag16& a, const Frag16& b, v8f c) {
  return __builtin_amdgcn_wmma_f32_16x16x32_bf16(
      false, a.v, false, b.v, (short)0, c, false, false);
}

// Low 32 bits of a generic pointer into LDS == wave-relative LDS byte offset.
__device__ __forceinline__ unsigned lds_off(const void* p) {
  return (unsigned)(size_t)p;
}

// CDNA5 async global->LDS copy (16B per lane), tracked by ASYNCcnt.
__device__ __forceinline__ void async_ld128(const unsigned short* g, unsigned lds) {
  asm volatile("global_load_async_to_lds_b128 %0, %1, off"
               :: "v"(lds), "v"(g) : "memory");
}
__device__ __forceinline__ void wait_async0() {
  asm volatile("s_wait_asynccnt 0x0" ::: "memory");
}

// ---------------------------------------------------------------------------
// fp32 -> bf16 elementwise conversion (vectorized x4)
// ---------------------------------------------------------------------------
__global__ void cvt_f32_bf16(const float* __restrict__ src,
                             unsigned short* __restrict__ dst, int n4) {
  int i = blockIdx.x * blockDim.x + threadIdx.x;
  if (i < n4) {
    float4 f = ((const float4*)src)[i];
    unsigned int lo = (unsigned int)f2bf(f.x) | ((unsigned int)f2bf(f.y) << 16);
    unsigned int hi = (unsigned int)f2bf(f.z) | ((unsigned int)f2bf(f.w) << 16);
    ((uint2*)dst)[i] = make_uint2(lo, hi);
  }
}

// ---------------------------------------------------------------------------
// GEMM core: out = X(bf16)[8192,1024] @ W(bf16)[1024,1024]^T, 64x64 tile,
// K-step 32, double-buffered async LDS staging.  W^T column-major == W
// row-major, so staging W rows gives the N-major WMMA B layout for free.
// ---------------------------------------------------------------------------
#define BM 64
#define BN 64
#define BK 32
#define LSTR 40   // padded LDS row stride in bf16 elems (80B: 16B-aligned, 20 banks)

// grid (16, 128, 3): z selects Q/K/V. Output written bf16 in [B,H,S,D].
__global__ __launch_bounds__(128) void gemm_qkv(
    const unsigned short* __restrict__ xb,
    const unsigned short* __restrict__ wq, const unsigned short* __restrict__ wk,
    const unsigned short* __restrict__ wv,
    const float* __restrict__ bq, const float* __restrict__ bk,
    const float* __restrict__ bv,
    unsigned short* __restrict__ qo, unsigned short* __restrict__ ko,
    unsigned short* __restrict__ vo) {
  __shared__ unsigned short Als[2][BM * LSTR];
  __shared__ unsigned short Bls[2][BN * LSTR];

  const unsigned short* W; const float* bias; unsigned short* out;
  if (blockIdx.z == 0)      { W = wq; bias = bq; out = qo; }
  else if (blockIdx.z == 1) { W = wk; bias = bk; out = ko; }
  else                      { W = wv; bias = bv; out = vo; }

  const int tid = threadIdx.x;
  const int lane = tid & 31, w = tid >> 5;
  const int hi = lane >> 4, ln = lane & 15;
  const int tileM = blockIdx.y * BM;
  const int tileN = blockIdx.x * BN;

  // 64x32 tile = 128 chunks of 16 elems; one 32B row-segment per thread.
  const int lr = tid >> 1;            // 0..63
  const int lo = (tid & 1) * 16;      // 0 or 16

  auto stage = [&](int buf, int kk) {
    const unsigned short* ga = xb + (size_t)(tileM + lr) * EMBED + kk + lo;
    unsigned la = lds_off(&Als[buf][lr * LSTR + lo]);
    async_ld128(ga, la);
    async_ld128(ga + 8, la + 16);
    const unsigned short* gb = W + (size_t)(tileN + lr) * EMBED + kk + lo;
    unsigned lb = lds_off(&Bls[buf][lr * LSTR + lo]);
    async_ld128(gb, lb);
    async_ld128(gb + 8, lb + 16);
  };

  v8f acc[4] = {v8f{}, v8f{}, v8f{}, v8f{}};

  stage(0, 0);
  for (int kk = 0; kk < EMBED; kk += BK) {
    const int buf = (kk >> 5) & 1;
    wait_async0();          // our tile-kk async copies have landed in LDS
    __syncthreads();        // everyone's have; previous buffer free to reuse
    if (kk + BK < EMBED) stage(buf ^ 1, kk + BK);

    // Gather A and all four B fragments first, then issue WMMAs
    // back-to-back (single dscnt wait, full XDL pipelining).
    Frag16 a;
#pragma unroll
    for (int v = 0; v < 8; ++v)
      a.u[v] = *(const unsigned int*)(&Als[buf][(w * 16 + ln) * LSTR + a_kbase(v, hi)]);

    Frag16 b[4];
#pragma unroll
    for (int nt = 0; nt < 4; ++nt)
#pragma unroll
      for (int v = 0; v < 8; ++v)
        b[nt].u[v] = *(const unsigned int*)(&Bls[buf][(nt * 16 + ln) * LSTR + a_kbase(v, hi)]);

#pragma unroll
    for (int nt = 0; nt < 4; ++nt)
      acc[nt] = wmma_bf16(a, b[nt], acc[nt]);
  }

  // Epilogue: +bias, cast bf16, permuted store into [B,H,S,D].
#pragma unroll
  for (int nt = 0; nt < 4; ++nt) {
    int c = tileN + nt * 16 + ln;
    float bb = bias[c];
    int h = c >> 6, d = c & 63;
#pragma unroll
    for (int j = 0; j < 8; ++j) {
      int r = tileM + w * 16 + hi * 8 + j;      // token index
      int bat = r >> 11, s = r & 2047;
      size_t o = (((size_t)(bat * HEADS + h) * SEQ + s) * HDIM + d);
      out[o] = f2bf(acc[nt][j] + bb);
    }
  }
}

// ---------------------------------------------------------------------------
// Flash attention: grid (32 q-tiles, 64 b*h), 128 threads (4 waves x 16 rows).
// Online softmax over 64 key-tiles of 32; Q lives in registers all loop.
// K/V tiles double-buffered via async global->LDS.
// ---------------------------------------------------------------------------
#define KSTR 72   // 32x64 K/V tiles, padded stride (144B row: aligned, 36 banks)
#define PSTR 36   // per-wave 16x32 P staging

__global__ __launch_bounds__(128) void attn_flash(
    const unsigned short* __restrict__ Q, const unsigned short* __restrict__ K,
    const unsigned short* __restrict__ V, unsigned short* __restrict__ O) {
  __shared__ unsigned short Kls[2][32 * KSTR];
  __shared__ unsigned short Vls[2][32 * KSTR];
  __shared__ unsigned short Pls[4 * 16 * PSTR];

  const int tid = threadIdx.x;
  const int lane = tid & 31, w = tid >> 5;
  const int hi = lane >> 4, ln = lane & 15;
  const int bh = blockIdx.y;
  const int qrow = blockIdx.x * 64 + w * 16 + ln;   // A-layout row M = ln

  // 32x64 tile = one 32B row-segment per thread.
  const int kr = tid >> 2;            // 0..31 (key row)
  const int ko = (tid & 3) * 16;      // 0,16,32,48 (dim offset)

  auto stage = [&](int buf, int kt) {
    size_t g = ((size_t)bh * SEQ + kt * 32 + kr) * HDIM + ko;
    unsigned lk = lds_off(&Kls[buf][kr * KSTR + ko]);
    async_ld128(K + g, lk);
    async_ld128(K + g + 8, lk + 16);
    unsigned lv = lds_off(&Vls[buf][kr * KSTR + ko]);
    async_ld128(V + g, lv);
    async_ld128(V + g + 8, lv + 16);
  };

  // Q A-fragments for dims [0,32) and [32,64) -- loaded once.
  const unsigned short* qp = Q + ((size_t)bh * SEQ + qrow) * HDIM;
  Frag16 qa[2];
#pragma unroll
  for (int f = 0; f < 2; ++f)
#pragma unroll
    for (int v = 0; v < 8; ++v)
      qa[f].u[v] = *(const unsigned int*)(qp + f * 32 + a_kbase(v, hi));

  float m[8], l[8];
#pragma unroll
  for (int j = 0; j < 8; ++j) { m[j] = -1e30f; l[j] = 0.0f; }
  v8f o[4] = {v8f{}, v8f{}, v8f{}, v8f{}};

  const float scale = 0.125f;   // 1/sqrt(64)

  stage(0, 0);
  for (int kt = 0; kt < SEQ / 32; ++kt) {
    const int buf = kt & 1;
    wait_async0();
    __syncthreads();
    if (kt + 1 < SEQ / 32) stage(buf ^ 1, kt + 1);

    // Scores: two 16x16 tiles (keys sub*16..+15), K-dim 64 in 2 WMMAs each.
    // Pre-gather all four K B-fragments, then 4 WMMAs back-to-back
    // (two independent accumulation chains).
    Frag16 bk[2][2];   // [sub][f]
#pragma unroll
    for (int sub = 0; sub < 2; ++sub)
#pragma unroll
      for (int f = 0; f < 2; ++f)
#pragma unroll
        for (int v = 0; v < 8; ++v)
          bk[sub][f].u[v] = *(const unsigned int*)(
              &Kls[buf][(sub * 16 + ln) * KSTR + f * 32 + a_kbase(v, hi)]);

    v8f s[2];
#pragma unroll
    for (int sub = 0; sub < 2; ++sub) {
      v8f a = {};
      a = wmma_bf16(qa[0], bk[sub][0], a);
      a = wmma_bf16(qa[1], bk[sub][1], a);
#pragma unroll
      for (int j = 0; j < 8; ++j) a[j] *= scale;
      s[sub] = a;
    }

    // Online softmax (row j lives across the 16 lanes of this half-wave).
#pragma unroll
    for (int j = 0; j < 8; ++j) {
      float t = fmaxf(s[0][j], s[1][j]);
#pragma unroll
      for (int off = 8; off > 0; off >>= 1) t = fmaxf(t, __shfl_xor(t, off, 16));
      float mn = fmaxf(m[j], t);
      float cf = __expf(m[j] - mn);
      m[j] = mn;
      float p0 = __expf(s[0][j] - mn);
      float p1 = __expf(s[1][j] - mn);
      s[0][j] = p0; s[1][j] = p1;
      float rs = p0 + p1;
#pragma unroll
      for (int off = 8; off > 0; off >>= 1) rs += __shfl_xor(rs, off, 16);
      l[j] = l[j] * cf + rs;
#pragma unroll
      for (int dt = 0; dt < 4; ++dt) o[dt][j] *= cf;
    }

    // C-layout P -> A-layout P via per-wave LDS staging (bf16).
    unsigned short* pb = &Pls[w * 16 * PSTR];
#pragma unroll
    for (int sub = 0; sub < 2; ++sub)
#pragma unroll
      for (int j = 0; j < 8; ++j)
        pb[(hi * 8 + j) * PSTR + sub * 16 + ln] = f2bf(s[sub][j]);
    // same-wave LDS is in-order: read back immediately
    Frag16 pa;
#pragma unroll
    for (int v = 0; v < 8; ++v)
      pa.u[v] = *(const unsigned int*)(&pb[ln * PSTR + a_kbase(v, hi)]);

    // O += P(16x32) x V(32x64): pre-gather all four V^T B-fragments,
    // then 4 WMMAs back-to-back.
    Frag16 bv[4];
#pragma unroll
    for (int dt = 0; dt < 4; ++dt)
#pragma unroll
      for (int v = 0; v < 8; ++v) {
        int key = a_kbase(v, hi);
        int dim = dt * 16 + ln;
        unsigned int e0 = Vls[buf][key * KSTR + dim];
        unsigned int e1 = Vls[buf][(key + 1) * KSTR + dim];
        bv[dt].u[v] = e0 | (e1 << 16);
      }
#pragma unroll
    for (int dt = 0; dt < 4; ++dt)
      o[dt] = wmma_bf16(pa, bv[dt], o[dt]);
  }

  // Normalize and store bf16 into [tok, E] with col = h*64+d.
  const int bat = bh >> 4, h = bh & 15;
#pragma unroll
  for (int j = 0; j < 8; ++j) {
    float inv = 1.0f / l[j];
    int srow = blockIdx.x * 64 + w * 16 + hi * 8 + j;
    size_t tok = (size_t)bat * SEQ + srow;
#pragma unroll
    for (int dt = 0; dt < 4; ++dt) {
      int col = h * 64 + dt * 16 + ln;
      O[tok * EMBED + col] = f2bf(o[dt][j] * inv);
    }
  }
}

// ---------------------------------------------------------------------------
// Output projection: out(f32) = attn(bf16) @ Wo^T + bo
// ---------------------------------------------------------------------------
__global__ __launch_bounds__(128) void gemm_out(
    const unsigned short* __restrict__ ab, const unsigned short* __restrict__ wo,
    const float* __restrict__ bo, float* __restrict__ out) {
  __shared__ unsigned short Als[2][BM * LSTR];
  __shared__ unsigned short Bls[2][BN * LSTR];

  const int tid = threadIdx.x;
  const int lane = tid & 31, w = tid >> 5;
  const int hi = lane >> 4, ln = lane & 15;
  const int tileM = blockIdx.y * BM;
  const int tileN = blockIdx.x * BN;

  const int lr = tid >> 1;
  const int lo = (tid & 1) * 16;

  auto stage = [&](int buf, int kk) {
    const unsigned short* ga = ab + (size_t)(tileM + lr) * EMBED + kk + lo;
    unsigned la = lds_off(&Als[buf][lr * LSTR + lo]);
    async_ld128(ga, la);
    async_ld128(ga + 8, la + 16);
    const unsigned short* gb = wo + (size_t)(tileN + lr) * EMBED + kk + lo;
    unsigned lb = lds_off(&Bls[buf][lr * LSTR + lo]);
    async_ld128(gb, lb);
    async_ld128(gb + 8, lb + 16);
  };

  v8f acc[4] = {v8f{}, v8f{}, v8f{}, v8f{}};

  stage(0, 0);
  for (int kk = 0; kk < EMBED; kk += BK) {
    const int buf = (kk >> 5) & 1;
    wait_async0();
    __syncthreads();
    if (kk + BK < EMBED) stage(buf ^ 1, kk + BK);

    Frag16 a;
#pragma unroll
    for (int v = 0; v < 8; ++v)
      a.u[v] = *(const unsigned int*)(&Als[buf][(w * 16 + ln) * LSTR + a_kbase(v, hi)]);

    Frag16 b[4];
#pragma unroll
    for (int nt = 0; nt < 4; ++nt)
#pragma unroll
      for (int v = 0; v < 8; ++v)
        b[nt].u[v] = *(const unsigned int*)(&Bls[buf][(nt * 16 + ln) * LSTR + a_kbase(v, hi)]);

#pragma unroll
    for (int nt = 0; nt < 4; ++nt)
      acc[nt] = wmma_bf16(a, b[nt], acc[nt]);
  }

#pragma unroll
  for (int nt = 0; nt < 4; ++nt) {
    int c = tileN + nt * 16 + ln;
    float bb = bo[c];
#pragma unroll
    for (int j = 0; j < 8; ++j) {
      int r = tileM + w * 16 + hi * 8 + j;
      out[(size_t)r * EMBED + c] = acc[nt][j] + bb;
    }
  }
}

// ---------------------------------------------------------------------------
// Launch.  Workspace layout (bf16 halves, 256B aligned by construction):
//   xb 16MiB | Wq/Wk/Wv/Wo 2MiB each | Q/K/V 16MiB each | attn 16MiB = 88MiB
// ---------------------------------------------------------------------------
extern "C" void kernel_launch(void* const* d_in, const int* in_sizes, int n_in,
                              void* d_out, int out_size, void* d_ws, size_t ws_size,
                              hipStream_t stream) {
  const float* x  = (const float*)d_in[0];
  const float* Wq = (const float*)d_in[1];
  const float* bq = (const float*)d_in[2];
  const float* Wk = (const float*)d_in[3];
  const float* bk = (const float*)d_in[4];
  const float* Wv = (const float*)d_in[5];
  const float* bv = (const float*)d_in[6];
  const float* Wo = (const float*)d_in[7];
  const float* bo = (const float*)d_in[8];
  float* out = (float*)d_out;

  unsigned short* xb  = (unsigned short*)d_ws;
  unsigned short* wqb = xb  + (size_t)TOKENS * EMBED;
  unsigned short* wkb = wqb + (size_t)EMBED * EMBED;
  unsigned short* wvb = wkb + (size_t)EMBED * EMBED;
  unsigned short* wob = wvb + (size_t)EMBED * EMBED;
  unsigned short* qb  = wob + (size_t)EMBED * EMBED;
  unsigned short* kb  = qb  + (size_t)TOKENS * EMBED;
  unsigned short* vb  = kb  + (size_t)TOKENS * EMBED;
  unsigned short* ab  = vb  + (size_t)TOKENS * EMBED;

  // fp32 -> bf16 conversions
  {
    int n4 = TOKENS * EMBED / 4;
    cvt_f32_bf16<<<(n4 + 255) / 256, 256, 0, stream>>>(x, xb, n4);
    int w4 = EMBED * EMBED / 4;
    cvt_f32_bf16<<<(w4 + 255) / 256, 256, 0, stream>>>(Wq, wqb, w4);
    cvt_f32_bf16<<<(w4 + 255) / 256, 256, 0, stream>>>(Wk, wkb, w4);
    cvt_f32_bf16<<<(w4 + 255) / 256, 256, 0, stream>>>(Wv, wvb, w4);
    cvt_f32_bf16<<<(w4 + 255) / 256, 256, 0, stream>>>(Wo, wob, w4);
  }

  // Q/K/V projections (z selects matrix)
  gemm_qkv<<<dim3(EMBED / BN, TOKENS / BM, 3), 128, 0, stream>>>(
      xb, wqb, wkb, wvb, bq, bk, bv, qb, kb, vb);

  // Flash attention
  attn_flash<<<dim3(SEQ / 64, BATCH * HEADS), 128, 0, stream>>>(qb, kb, vb, ab);

  // Output projection
  gemm_out<<<dim3(EMBED / BN, TOKENS / BM), 128, 0, stream>>>(ab, wob, bo, out);
}
